// NeuralODE_60181081751902
// MI455X (gfx1250) — matrix operand order repaired and checked by reading, not compiled, not run on hardware
//
#include <hip/hip_runtime.h>
#include <hip/hip_bf16.h>
#include <math.h>

typedef _Float16 v16h __attribute__((ext_vector_type(16)));
typedef _Float16 v8h  __attribute__((ext_vector_type(8)));
typedef float    v8f  __attribute__((ext_vector_type(8)));

// D(16x16 f32) = A(16x32 f16) * B(32x16 f16) + C(16x16 f32), wave32
#define WMMA32_F16(A, B, C) \
  __builtin_amdgcn_wmma_f32_16x16x32_f16(false, (A), false, (B), (short)0, (C), false, false)

// Compiler-only memory fence: zero instructions. Hardware ordering is free
// because same-wave LDS ops execute in order in the DS pipe (ISA 7.3), and
// all cross-lane traffic here is within a single wave.
#define WAVE_LDS_FENCE() asm volatile("" ::: "memory")

// gfx1250 hardware tanh (TRANS pipe).
__device__ __forceinline__ float fast_tanh(float v) {
#if __has_builtin(__builtin_amdgcn_tanhf)
  return __builtin_amdgcn_tanhf(v);
#else
  const float a = __builtin_fabsf(v);
  const float e = __builtin_amdgcn_exp2f(-2.8853900817779268f * a); // exp(-2a)
  const float r = (1.0f - e) * __builtin_amdgcn_rcpf(1.0f + e);
  return __builtin_copysignf(r, v);
#endif
}

// Pack two f32 into one dword of f16 (v_cvt_pk_rtz_f16_f32).
__device__ __forceinline__ unsigned int pk2(float a, float b) {
  typedef __fp16 f16x2 __attribute__((ext_vector_type(2)));
  f16x2 p = __builtin_amdgcn_cvt_pkrtz(a, b);
  return __builtin_bit_cast(unsigned int, p);
}

namespace {
constexpr int Bsz = 4096;
constexpr int S   = 512;
constexpr int D   = 8;     // state dim
constexpr int H   = 128;   // hidden dim
}

// One wave (32 lanes) integrates a 16-row batch tile through all S-1 Euler
// steps. Mixed precision: f32 state / accumulate / tanh; f16 GEMM inputs.
// Transposed dataflow: hT = W1T @ xT, yT = W2T @ hT, so the constant weights
// are the A operands and per-step data are B operands.
//
// f16 layouts as used here (same K order for A and B, wave32), hlf = lane/16, c = lane%16:
//   A 16x32 frag (v16h): row M = c; half h -> K = h + 16*hlf
//   B 32x16 frag (v16h): col N = c; half h -> K = h + 16*hlf
//     => halves are 2 contiguous 16B runs in K-major-per-column staging
//   C/D 16x16 f32 (v8f): vgpr r -> row M = r + 8*hlf, col N = c
__global__ __launch_bounds__(32) void node_euler_wmma(
    const float* __restrict__ x0,    // [B, D]
    const float* __restrict__ cur,   // [B, S, 1]
    const float* __restrict__ t,     // [S]
    const float* __restrict__ W1,    // [D+1, H]
    const float* __restrict__ b1,    // [H]
    const float* __restrict__ W2,    // [H, D]
    const float* __restrict__ b2,    // [D]
    float* __restrict__ traj)        // [B, S, D]
{
  // Batch-major f16 staging (row = batch column of the transposed GEMMs).
  __shared__ __align__(16) _Float16 xbuf[16][32];  // [batch][K]: x(0..7), cur(8), 1.0(9), 0(10..31)
  __shared__ __align__(16) _Float16 hbuf[16][H];   // [batch][hidden]

  const int lane = threadIdx.x;      // 0..31
  const int hlf  = lane >> 4;        // lane half (0/1)
  const int c    = lane & 15;        // batch column (B/C layouts) / A row M
  const int b0   = blockIdx.x * 16;  // batch tile base

  // ---- W1T A-fragments (constant), bias b1 folded in at K==9 ----
  v16h w1a[8];
#pragma unroll
  for (int n = 0; n < 8; ++n)
#pragma unroll
    for (int h = 0; h < 16; ++h) {
      const int K = h + 16 * hlf;  // 0..31
      const int hid = n * 16 + c;
      w1a[n][h] = (K <= D) ? (_Float16)W1[K * H + hid]
                : (K == D + 1) ? (_Float16)b1[hid]
                : (_Float16)0.0f;
    }

  // ---- W2T A-fragments (constant), out-feature rows 8..15 zero-padded ----
  v16h w2a[4];
#pragma unroll
  for (int k = 0; k < 4; ++k)
#pragma unroll
    for (int h = 0; h < 16; ++h) {
      const int K = 32 * k + (h + 16 * hlf); // 0..127
      w2a[k][h] = (c < D) ? (_Float16)W2[K * D + c] : (_Float16)0.0f;
    }

  // ---- b2 bias in C-layout (row = out feature = r + 8*hlf) ----
  v8f bb2;
#pragma unroll
  for (int r = 0; r < 8; ++r) bb2[r] = (hlf == 0) ? b2[r] : 0.0f;

  // ---- x state: lane c of half 0 owns batch row b0+c (8 f32, contiguous) ----
  v8f x = {};
  if (hlf == 0) {
    x = *(const v8f*)(x0 + (size_t)(b0 + c) * D);
    *(v8f*)(traj + (size_t)(b0 + c) * S * D) = x;  // traj[:,0,:] = x0
  }

  // Zero all of xbuf once, then plant the constant-1 bias input at K==9.
  {
    v8h z = {};
    _Float16* p = &xbuf[0][0];
    *(v8h*)(p + lane * 16) = z;          // 32 lanes x 16 halves = whole xbuf
    *(v8h*)(p + lane * 16 + 8) = z;      // (second half of each 16-half run)
  }
  WAVE_LDS_FENCE();
  if (hlf == 0) xbuf[c][D + 1] = (_Float16)1.0f;

  // software-pipelined current-profile value (left endpoint of step s)
  float curv = (lane < 16) ? cur[(size_t)(b0 + lane) * S] : 0.0f;

#pragma unroll 1
  for (int s = 0; s < S - 1; ++s) {
    const float dt = t[s + 1] - t[s];

    // stage xT: one b128 (x as f16) + one b16 (cur); K=9 stays 1.0, K>=10 stays 0
    if (hlf == 0) {
      uint4 w;
      w.x = pk2(x[0], x[1]);
      w.y = pk2(x[2], x[3]);
      w.z = pk2(x[4], x[5]);
      w.w = pk2(x[6], x[7]);
      *(uint4*)&xbuf[c][0] = w;
      xbuf[c][D] = (_Float16)curv;
    }
    WAVE_LDS_FENCE();

    // prefetch next step's current value (overlaps this whole step)
    {
      const int sn = (s + 1 < S - 1) ? s + 1 : s;
      if (lane < 16) curv = cur[(size_t)(b0 + lane) * S + sn];
    }

    // B-fragment of xT (32x16): halves h -> K = h + 16*hlf at batch c
    v16h bx;
    {
      const v8h lo = *(const v8h*)&xbuf[c][16 * hlf];
      const v8h hi = *(const v8h*)&xbuf[c][16 * hlf + 8];
      bx = __builtin_shufflevector(lo, hi, 0, 1, 2, 3, 4, 5, 6, 7,
                                   8, 9, 10, 11, 12, 13, 14, 15);
    }

    // ---- GEMM1T: hT = tanh(W1T @ xT), bias via constant-1 input ----
    // Phase A: issue all 8 independent WMMAs (distinct live D ranges =>
    // back-to-back XDL pipelining, no per-tile hazard stalls).
    v8f acc[8];
#pragma unroll
    for (int n = 0; n < 8; ++n) {
      v8f z = {};                                 // inline-0 C operand
      acc[n] = WMMA32_F16(w1a[n], bx, z);
    }
    // Phase B: tanh + pack + contiguous b128 stores.
#pragma unroll
    for (int n = 0; n < 8; ++n) {
      uint4 w;
      w.x = pk2(fast_tanh(acc[n][0]), fast_tanh(acc[n][1]));
      w.y = pk2(fast_tanh(acc[n][2]), fast_tanh(acc[n][3]));
      w.z = pk2(fast_tanh(acc[n][4]), fast_tanh(acc[n][5]));
      w.w = pk2(fast_tanh(acc[n][6]), fast_tanh(acc[n][7]));
      *(uint4*)&hbuf[c][n * 16 + 8 * hlf] = w;
    }
    WAVE_LDS_FENCE();

    // ---- GEMM2T: yT = W2T @ hT + b2; 4 fully independent WMMAs ----
    v8f y[4];
#pragma unroll
    for (int k = 0; k < 4; ++k) {
      v16h bh;
      const v8h lo = *(const v8h*)&hbuf[c][32 * k + 16 * hlf];
      const v8h hi = *(const v8h*)&hbuf[c][32 * k + 16 * hlf + 8];
      bh = __builtin_shufflevector(lo, hi, 0, 1, 2, 3, 4, 5, 6, 7,
                                   8, 9, 10, 11, 12, 13, 14, 15);
      if (k == 0) {
        y[k] = WMMA32_F16(w2a[k], bh, bb2);       // bias in chain 0
      } else {
        v8f z = {};
        y[k] = WMMA32_F16(w2a[k], bh, z);
      }
    }
    const v8f yt = (y[0] + y[1]) + (y[2] + y[3]); // rows 8..15 (half 1) stay 0

    // ---- Euler update (f32) + one contiguous 32B trajectory store ----
#pragma unroll
    for (int r = 0; r < 8; ++r) x[r] = x[r] + dt * yt[r];
    if (hlf == 0)
      *(v8f*)(traj + (size_t)(b0 + c) * S * D + (size_t)(s + 1) * D) = x;
    WAVE_LDS_FENCE();  // keep next iteration's LDS writes behind this step's reads
  }
}

extern "C" void kernel_launch(void* const* d_in, const int* in_sizes, int n_in,
                              void* d_out, int out_size, void* d_ws, size_t ws_size,
                              hipStream_t stream) {
  (void)in_sizes; (void)n_in; (void)out_size; (void)d_ws; (void)ws_size;
  const float* x0 = (const float*)d_in[0];
  const float* cp = (const float*)d_in[1];
  const float* t  = (const float*)d_in[2];
  const float* W1 = (const float*)d_in[3];
  const float* b1 = (const float*)d_in[4];
  const float* W2 = (const float*)d_in[5];
  const float* b2 = (const float*)d_in[6];
  float* traj = (float*)d_out;

  dim3 grid(Bsz / 16);   // 256 independent 16-row batch tiles
  dim3 block(32);        // one wave32 per tile
  node_euler_wmma<<<grid, block, 0, stream>>>(x0, cp, t, W1, b1, W2, b2, traj);
}
